// CAVAE_dagnn_41601053229538
// MI455X (gfx1250) — matrix operand order repaired and checked by reading, not compiled, not run on hardware
//
#include <hip/hip_runtime.h>
#include <hip/hip_bf16.h>

// ---------------------------------------------------------------------------
// CAVAE DAG-GNN for MI455X (gfx1250, wave32, WMMA f32_16x16x32_f16)
//
//  * solve(I, X) == X (I identity) -> b_inv = I - adjB, no solver
//  * scan keeps P0 = buf@Wr0^T, P1 = buf@Wr1^T, kbuf = buf.wk incrementally
//  * GRU input halves depending only on Hl hoisted into full GEMMs
//  * 96-step scan = ONE persistent 768-thread block (24 waves)
//  * big GEMMs: LDS double-buffered via global_load_async_to_lds_b128
//    (ASYNCcnt DMA path), WMMA fed from ds_load_b128
// ---------------------------------------------------------------------------

#define BB    8
#define NN    96
#define EMB   768
#define HID   384
#define CODE  192
#define FEAT  768
#define H3    1152   // 3*HID
#define ROWS  768    // B*N

typedef _Float16 half_t;
typedef __attribute__((ext_vector_type(8)))  _Float16 v8h;
typedef __attribute__((ext_vector_type(16))) _Float16 v16h;
typedef __attribute__((ext_vector_type(8)))  float    v8f;

// ---------------- CDNA5 async global->LDS DMA ------------------------------
// LDS generic pointers: low 32 bits are the LDS byte offset (aperture in
// high bits), so the raw u32 works as the VDST LDS address.
__device__ __forceinline__ void async_b128(const void* g, const half_t* lds) {
  unsigned l = (unsigned)(uintptr_t)lds;
  asm volatile("global_load_async_to_lds_b128 %0, %1, off"
               :: "v"(l), "v"(g) : "memory");
}
template <int N>
__device__ __forceinline__ void wait_asynccnt() {
  asm volatile("s_wait_asynccnt %0" :: "n"(N) : "memory");
}

// ---------------- WMMA fragment helpers (CDNA5 16x16x32 f16 layout) --------
__device__ __forceinline__ v16h load_a_frag(const half_t* A, int lda) {
  int lane = threadIdx.x & 31;
  const half_t* p = A + (lane & 15) * lda + ((lane >> 4) << 3);
  v8h lo = *(const v8h*)(p);
  v8h hi = *(const v8h*)(p + 16);
  return __builtin_shufflevector(lo, hi, 0,1,2,3,4,5,6,7,8,9,10,11,12,13,14,15);
}
__device__ __forceinline__ v16h load_b_frag(const half_t* W, int ldw) {
  int lane = threadIdx.x & 31;
  const half_t* p = W + (lane & 15) * ldw + ((lane >> 4) << 3);
  v8h lo = *(const v8h*)(p);
  v8h hi = *(const v8h*)(p + 16);
  return __builtin_shufflevector(lo, hi, 0,1,2,3,4,5,6,7,8,9,10,11,12,13,14,15);
}
__device__ __forceinline__ v8f wmma32(v16h a, v16h b, v8f c) {
  return __builtin_amdgcn_wmma_f32_16x16x32_f16(false, a, false, b, (short)0, c,
                                                false, false);
}
__device__ __forceinline__ float sigmoidf_(float x) {
  return 1.0f / (1.0f + __expf(-x));
}

// ---------------- GEMM: C = act(A @ W^T + bias) ----------------------------
// Block = 256 threads (8 waves), 128x64 output tile. K staged 32 at a time
// through double-buffered LDS via async DMA. Rows padded to 40 halfs.
#define APAD 40
__global__ void __launch_bounds__(256)
gemm_f16(const half_t* __restrict__ A, int lda,
         const half_t* __restrict__ W, int ldw,
         const float* __restrict__ bias,
         float* __restrict__ C, int ldc,
         half_t* __restrict__ Ch, int ldch,
         int M, int N, int K, int relu)
{
  __shared__ __align__(16) half_t Ah[2][128][APAD];
  __shared__ __align__(16) half_t Bh[2][64][APAD];

  const int tid  = threadIdx.x;
  const int lane = tid & 31;
  const int wv   = tid >> 5;            // 0..7 -> 16-row strip
  const int nblk = N >> 6;
  const int bm   = blockIdx.x / nblk;   // 128-row block
  const int bn   = blockIdx.x % nblk;   // 64-col block

  // stage one 32-K slice into buffer `buf` (3 async b128 per thread)
  auto stage = [&](int buf, int k0) {
    #pragma unroll
    for (int c = tid; c < 512; c += 256) {          // A: 128 rows x 4 chunks
      int row = c >> 2, kc = c & 3;
      async_b128(A + (size_t)(bm * 128 + row) * lda + k0 + kc * 8,
                 &Ah[buf][row][kc * 8]);
    }
    {                                                // B: 64 rows x 4 chunks
      int row = tid >> 2, kc = tid & 3;
      async_b128(W + (size_t)(bn * 64 + row) * ldw + k0 + kc * 8,
                 &Bh[buf][row][kc * 8]);
    }
  };

  const int KT = K >> 5;
  v8f acc[4] = {v8f{}, v8f{}, v8f{}, v8f{}};

  stage(0, 0);
  for (int kt = 0; kt < KT; ++kt) {
    int cur = kt & 1;
    if (kt + 1 < KT) {
      stage(1 - cur, (kt + 1) << 5);
      wait_asynccnt<3>();               // previous stage done (in-order)
    } else {
      wait_asynccnt<0>();
    }
    __syncthreads();
    v16h a = load_a_frag(&Ah[cur][wv * 16][0], APAD);
    #pragma unroll
    for (int s = 0; s < 4; ++s) {
      v16h b = load_b_frag(&Bh[cur][s * 16][0], APAD);
      acc[s] = wmma32(a, b, acc[s]);
    }
    __syncthreads();                    // buf reusable after all waves read
  }

  int mo = (lane >> 4) * 8;
  int cn = lane & 15;
  #pragma unroll
  for (int s = 0; s < 4; ++s) {
    int col = bn * 64 + s * 16 + cn;
    float bz = bias ? bias[col] : 0.0f;
    #pragma unroll
    for (int g = 0; g < 8; ++g) {
      int m = bm * 128 + wv * 16 + mo + g;
      float v = acc[s][g] + bz;
      if (relu) v = fmaxf(v, 0.0f);
      C[(size_t)m * ldc + col] = v;
      if (Ch) Ch[(size_t)m * ldch + col] = (half_t)v;
    }
  }
}

// ---------------- f32 -> f16 convert ---------------------------------------
__global__ void f32_to_f16(const float* __restrict__ s, half_t* __restrict__ d, int n) {
  int i = blockIdx.x * blockDim.x + threadIdx.x;
  if (i < n) d[i] = (half_t)s[i];
}

// ---------------- persistent DAG-scan kernel (one block, 24 waves) ---------
__global__ void __launch_bounds__(768)
scan_kernel(const float* __restrict__ Hl_f, const half_t* __restrict__ Hl_h, int ldH,
            float* __restrict__ Ho_f, half_t* __restrict__ Ho_h, int ldHo,
            const float* __restrict__ Gqc,   // (ROWS,H3) q@Wih_c^T+bih_c
            const float* __restrict__ Gqp,   // (ROWS,H3) q@Whh_p^T+bhh_p
            const float* __restrict__ bhh_c, // (H3)
            const float* __restrict__ bih_p, // (H3)
            const half_t* __restrict__ Whh_c,// (H3,HID)
            const half_t* __restrict__ Wih_p,// (H3,HID)
            const half_t* __restrict__ Wr0,  // (HID,HID)
            const half_t* __restrict__ Wr1,  // (HID,HID)
            float* __restrict__ P0, float* __restrict__ P1, // (B,N,HID)
            float* __restrict__ kbuf,        // (B,N) enc only
            const float* __restrict__ gatw,  // enc: wq|wk (768)
            const float* __restrict__ gatb,  // enc: scalar
            const float* __restrict__ adj,   // (B,N,N) enc only
            const float* __restrict__ smask, // (B,N,N)
            const float* __restrict__ binv_src, // dec: adjB (B,N,N)
            float* __restrict__ adjB_out,    // enc: (B,N,N)
            float* __restrict__ GhcG,        // scratch (BB,H3)
            float* __restrict__ GipG,        // scratch (BB,H3)
            int is_dec)
{
  __shared__ float   w_s[BB][NN];
  __shared__ float   qwq_s[BB];
  __shared__ float   M_s[BB][HID];
  __shared__ __align__(16) half_t Mh_s[16][HID];
  __shared__ float   hn_s[BB][HID];
  __shared__ __align__(16) half_t hnh_s[16][HID];

  const int tid  = threadIdx.x;
  const int lane = tid & 31;
  const int wv   = tid >> 5;         // 0..23

  for (int e = tid; e < 8 * HID; e += 768) {   // zero WMMA pad rows once
    Mh_s[8 + e / HID][e % HID]  = (half_t)0.0f;
    hnh_s[8 + e / HID][e % HID] = (half_t)0.0f;
  }
  __syncthreads();

  for (int i = 0; i < NN; ++i) {
    // ---- Phase A: attention weights w_s[b][n] -------------------------
    if (!is_dec) {
      if (wv < BB) {                                  // qwq[b] = q . wq
        const float* q = Hl_f + (size_t)(wv * NN + i) * ldH;
        float s = 0.0f;
        for (int d = lane; d < HID; d += 32) s += q[d] * gatw[d];
        for (int o = 16; o > 0; o >>= 1) s += __shfl_down(s, o, 32);
        if (lane == 0) qwq_s[wv] = s;
      }
      __syncthreads();
      float gb = *gatb;
      for (int e = tid; e < BB * NN; e += 768) {
        int b = e / NN, n = e % NN;
        float a;
        if (n < i) {
          a = kbuf[b * NN + n] + qwq_s[b] + gb;
          a -= (1.0f - adj[(size_t)(b * NN + i) * NN + n]) * 1.0e30f;
        } else {
          a = -__builtin_inff();
        }
        w_s[b][n] = a;
      }
      __syncthreads();
      if (tid < BB) {                                 // per-batch softmax
        int b = tid;
        if (i == 0) {
          for (int n = 0; n < NN; ++n) w_s[b][n] = 0.0f;
        } else {
          float mx = -__builtin_inff();
          for (int n = 0; n < i; ++n) mx = fmaxf(mx, w_s[b][n]);
          float sum = 0.0f;
          for (int n = 0; n < i; ++n) {
            float ev = __expf(w_s[b][n] - mx);
            w_s[b][n] = ev; sum += ev;
          }
          float inv = 1.0f / sum;
          for (int n = 0; n < i; ++n) w_s[b][n] *= inv;
          for (int n = i; n < NN; ++n) w_s[b][n] = 0.0f;
        }
      }
    } else {
      for (int e = tid; e < BB * NN; e += 768) {
        int b = e / NN, n = e % NN;
        w_s[b][n] = (n < i) ? -binv_src[(size_t)(b * NN + i) * NN + n] : 0.0f;
      }
    }
    __syncthreads();

    // ---- Phase B: M[b,d] = sum_n w * (s*P0 + (1-s)*P1) ----------------
    for (int e = tid; e < BB * HID; e += 768) {
      int b = e / HID, d = e % HID;
      const float* sm = smask + (size_t)(b * NN + i) * NN;
      const float* p0 = P0 + (size_t)b * NN * HID + d;
      const float* p1 = P1 + (size_t)b * NN * HID + d;
      float acc = 0.0f;
      for (int n = 0; n < i; ++n) {
        float wn = w_s[b][n];
        float sv = sm[n];
        acc += wn * (p0[(size_t)n * HID] * sv + p1[(size_t)n * HID] * (1.0f - sv));
      }
      M_s[b][d]  = acc;
      Mh_s[b][d] = (half_t)acc;
    }
    __syncthreads();

    // ---- Phase C: Ghc = M@Whh_c^T + bhh_c ; Gip = M@Wih_p^T + bih_p ---
    // 144 16x16 tiles over 24 waves; 1-deep register pipeline on (a,b)
    #pragma unroll
    for (int t = 0; t < 6; ++t) {
      int tile = wv * 6 + t;
      const half_t* Wm  = (tile < 72) ? Whh_c : Wih_p;
      const float* bvec = (tile < 72) ? bhh_c : bih_p;
      float* out        = (tile < 72) ? GhcG  : GipG;
      int ct = (tile < 72) ? tile : tile - 72;
      const half_t* Wt = Wm + (size_t)(ct * 16) * HID;
      v8f acc = {};
      v16h a = load_a_frag(&Mh_s[0][0], HID);
      v16h b = load_b_frag(Wt, HID);
      #pragma unroll
      for (int k = 0; k < HID; k += 32) {
        v16h a2 = a, b2 = b;
        if (k + 32 < HID) {
          a = load_a_frag(&Mh_s[0][k + 32], HID);
          b = load_b_frag(Wt + k + 32, HID);
        }
        acc = wmma32(a2, b2, acc);
      }
      int col = ct * 16 + (lane & 15);
      int mo  = (lane >> 4) * 8;
      #pragma unroll
      for (int g = 0; g < 8; ++g) {
        int m = mo + g;
        if (m < BB) out[(size_t)m * H3 + col] = acc[g] + bvec[col];
      }
    }
    __syncthreads();

    // ---- Phase D: dual-GRU elementwise, hn = gru_c(q,M) + gru_p(M,q) --
    for (int e = tid; e < BB * HID; e += 768) {
      int b = e / HID, d = e % HID;
      int r = b * NN + i;
      const float* gqc = Gqc + (size_t)r * H3;
      const float* gqp = Gqp + (size_t)r * H3;
      const float* ghc = GhcG + (size_t)b * H3;
      const float* gip = GipG + (size_t)b * H3;
      float m = M_s[b][d];
      float q = Hl_f[(size_t)r * ldH + d];
      float rc = sigmoidf_(gqc[d]           + ghc[d]);
      float zc = sigmoidf_(gqc[HID + d]     + ghc[HID + d]);
      float nc = tanhf    (gqc[2 * HID + d] + rc * ghc[2 * HID + d]);
      float oc = (1.0f - zc) * nc + zc * m;
      float rp = sigmoidf_(gip[d]           + gqp[d]);
      float zp = sigmoidf_(gip[HID + d]     + gqp[HID + d]);
      float np = tanhf    (gip[2 * HID + d] + rp * gqp[2 * HID + d]);
      float op = (1.0f - zp) * np + zp * q;
      float hn = oc + op;
      hn_s[b][d]  = hn;
      hnh_s[b][d] = (half_t)hn;
      Ho_f[(size_t)r * ldHo + d] = hn;
      Ho_h[(size_t)r * ldHo + d] = (half_t)hn;
    }
    if (!is_dec) {
      for (int e = tid; e < BB * NN; e += 768)
        adjB_out[(size_t)((e / NN) * NN + i) * NN + (e % NN)] = w_s[e / NN][e % NN];
    }
    __syncthreads();

    // ---- Phase E: P0/P1 row i = hn @ Wr{0,1}^T ; kbuf[i] = hn.wk ------
    if (!is_dec && wv < BB) {
      float s = 0.0f;
      for (int d = lane; d < HID; d += 32) s += hn_s[wv][d] * gatw[HID + d];
      for (int o = 16; o > 0; o >>= 1) s += __shfl_down(s, o, 32);
      if (lane == 0) kbuf[wv * NN + i] = s;
    }
    #pragma unroll
    for (int t = 0; t < 2; ++t) {
      int tile = wv * 2 + t;                  // 0..47
      const half_t* Wm = (tile < 24) ? Wr0 : Wr1;
      float* out       = (tile < 24) ? P0  : P1;
      int ct = tile % 24;
      const half_t* Wt = Wm + (size_t)(ct * 16) * HID;
      v8f acc = {};
      v16h a = load_a_frag(&hnh_s[0][0], HID);
      v16h b = load_b_frag(Wt, HID);
      #pragma unroll
      for (int k = 0; k < HID; k += 32) {
        v16h a2 = a, b2 = b;
        if (k + 32 < HID) {
          a = load_a_frag(&hnh_s[0][k + 32], HID);
          b = load_b_frag(Wt + k + 32, HID);
        }
        acc = wmma32(a2, b2, acc);
      }
      int col = ct * 16 + (lane & 15);
      int mo  = (lane >> 4) * 8;
      #pragma unroll
      for (int g = 0; g < 8; ++g) {
        int b2 = mo + g;
        if (b2 < BB) out[(size_t)(b2 * NN + i) * HID + col] = acc[g];
      }
    }
    __syncthreads();
  }
}

// ---------------- feature_map = I - adjB[1] --------------------------------
__global__ void feature_map_kernel(const float* __restrict__ adjB1,
                                   float* __restrict__ out) {
  int e = blockIdx.x * blockDim.x + threadIdx.x;
  if (e < BB * NN * NN) {
    int r = (e / NN) % NN, c = e % NN;
    out[e] = (r == c ? 1.0f : 0.0f) - adjB1[e];
  }
}

// ---------------------------------------------------------------------------
extern "C" void kernel_launch(void* const* d_in, const int* in_sizes, int n_in,
                              void* d_out, int out_size, void* d_ws, size_t ws_size,
                              hipStream_t stream) {
  const float* features   = (const float*)d_in[0];
  const float* adj        = (const float*)d_in[1];
  const float* s_mask     = (const float*)d_in[2];
  const float* enc_fc1_W  = (const float*)d_in[5];
  const float* enc_fc1_b  = (const float*)d_in[6];
  const float* enc_gat_w  = (const float*)d_in[7];
  const float* enc_gat_b  = (const float*)d_in[8];
  const float* enc_Wr0    = (const float*)d_in[9];
  const float* enc_Wr1    = (const float*)d_in[10];
  const float* enc_gruc_Wih = (const float*)d_in[11];
  const float* enc_gruc_Whh = (const float*)d_in[12];
  const float* enc_gruc_bih = (const float*)d_in[13];
  const float* enc_gruc_bhh = (const float*)d_in[14];
  const float* enc_grup_Wih = (const float*)d_in[15];
  const float* enc_grup_Whh = (const float*)d_in[16];
  const float* enc_grup_bih = (const float*)d_in[17];
  const float* enc_grup_bhh = (const float*)d_in[18];
  const float* enc_mlp_W0 = (const float*)d_in[19];
  const float* enc_mlp_b0 = (const float*)d_in[20];
  const float* enc_mlp_W1 = (const float*)d_in[21];
  const float* enc_mlp_b1 = (const float*)d_in[22];
  const float* dec_fc1_W  = (const float*)d_in[23];
  const float* dec_fc1_b  = (const float*)d_in[24];
  const float* dec_Wr0    = (const float*)d_in[25];
  const float* dec_Wr1    = (const float*)d_in[26];
  const float* dec_gruc_Wih = (const float*)d_in[27];
  const float* dec_gruc_Whh = (const float*)d_in[28];
  const float* dec_gruc_bih = (const float*)d_in[29];
  const float* dec_gruc_bhh = (const float*)d_in[30];
  const float* dec_grup_Wih = (const float*)d_in[31];
  const float* dec_grup_Whh = (const float*)d_in[32];
  const float* dec_grup_bih = (const float*)d_in[33];
  const float* dec_grup_bhh = (const float*)d_in[34];
  const float* dec_mlp_W0 = (const float*)d_in[35];
  const float* dec_mlp_b0 = (const float*)d_in[36];
  const float* dec_mlp_W1 = (const float*)d_in[37];
  const float* dec_mlp_b1 = (const float*)d_in[38];

  char* ws = (char*)d_ws;
  size_t off = 0;
  auto allocH = [&](size_t n) {
    half_t* p = (half_t*)(ws + off);
    off += ((n * sizeof(half_t) + 255) / 256) * 256; return p;
  };
  auto allocF = [&](size_t n) {
    float* p = (float*)(ws + off);
    off += ((n * 4 + 255) / 256) * 256; return p;
  };

  half_t* feat_h      = allocH((size_t)ROWS * EMB);
  half_t* encfc1W_h   = allocH((size_t)HID * EMB);
  half_t* encgcWih_h  = allocH((size_t)2 * H3 * HID);
  half_t* encgcWhh_h  = allocH((size_t)2 * H3 * HID);
  half_t* encgpWih_h  = allocH((size_t)2 * H3 * HID);
  half_t* encgpWhh_h  = allocH((size_t)2 * H3 * HID);
  half_t* encWr0_h    = allocH((size_t)2 * HID * HID);
  half_t* encWr1_h    = allocH((size_t)2 * HID * HID);
  half_t* encmlpW0_h  = allocH((size_t)HID * H3);
  half_t* encmlpW1_h  = allocH((size_t)CODE * HID);
  half_t* decfc1W_h   = allocH((size_t)HID * CODE);
  half_t* decgcWih_h  = allocH((size_t)2 * H3 * HID);
  half_t* decgcWhh_h  = allocH((size_t)2 * H3 * HID);
  half_t* decgpWih_h  = allocH((size_t)2 * H3 * HID);
  half_t* decgpWhh_h  = allocH((size_t)2 * H3 * HID);
  half_t* decWr0_h    = allocH((size_t)2 * HID * HID);
  half_t* decWr1_h    = allocH((size_t)2 * HID * HID);
  half_t* decmlpW0_h  = allocH((size_t)HID * H3);
  half_t* decmlpW1_h  = allocH((size_t)FEAT * HID);
  half_t* HcatEncH    = allocH((size_t)ROWS * H3);
  half_t* HcatDecH    = allocH((size_t)ROWS * H3);
  half_t* tmpH        = allocH((size_t)ROWS * HID);
  half_t* fUH         = allocH((size_t)ROWS * CODE);

  float* HcatEncF = allocF((size_t)ROWS * H3);
  float* HcatDecF = allocF((size_t)ROWS * H3);
  float* Gqc      = allocF((size_t)ROWS * H3);
  float* Gqp      = allocF((size_t)ROWS * H3);
  float* P0       = allocF((size_t)BB * NN * HID);
  float* P1       = allocF((size_t)BB * NN * HID);
  float* kbuf     = allocF((size_t)BB * NN);
  float* adjB0    = allocF((size_t)BB * NN * NN);
  float* adjB1    = allocF((size_t)BB * NN * NN);
  float* Ghc      = allocF((size_t)BB * H3);
  float* Gip      = allocF((size_t)BB * H3);
  float* tmpF     = allocF((size_t)ROWS * HID);
  float* fUF      = allocF((size_t)ROWS * CODE);

  auto conv = [&](const float* s, half_t* d, size_t n) {
    f32_to_f16<<<(int)((n + 255) / 256), 256, 0, stream>>>(s, d, (int)n);
  };
  auto gemm = [&](const half_t* A, int lda, const half_t* W, int ldw,
                  const float* bias, float* C, int ldc, half_t* Ch, int ldch,
                  int M, int N, int K, int relu) {
    int blocks = (M / 128) * (N / 64);
    gemm_f16<<<blocks, 256, 0, stream>>>(A, lda, W, ldw, bias,
                                         C, ldc, Ch, ldch, M, N, K, relu);
  };

  conv(features,     feat_h,     (size_t)ROWS * EMB);
  conv(enc_fc1_W,    encfc1W_h,  (size_t)HID * EMB);
  conv(enc_gruc_Wih, encgcWih_h, (size_t)2 * H3 * HID);
  conv(enc_gruc_Whh, encgcWhh_h, (size_t)2 * H3 * HID);
  conv(enc_grup_Wih, encgpWih_h, (size_t)2 * H3 * HID);
  conv(enc_grup_Whh, encgpWhh_h, (size_t)2 * H3 * HID);
  conv(enc_Wr0,      encWr0_h,   (size_t)2 * HID * HID);
  conv(enc_Wr1,      encWr1_h,   (size_t)2 * HID * HID);
  conv(enc_mlp_W0,   encmlpW0_h, (size_t)HID * H3);
  conv(enc_mlp_W1,   encmlpW1_h, (size_t)CODE * HID);
  conv(dec_fc1_W,    decfc1W_h,  (size_t)HID * CODE);
  conv(dec_gruc_Wih, decgcWih_h, (size_t)2 * H3 * HID);
  conv(dec_gruc_Whh, decgcWhh_h, (size_t)2 * H3 * HID);
  conv(dec_grup_Wih, decgpWih_h, (size_t)2 * H3 * HID);
  conv(dec_grup_Whh, decgpWhh_h, (size_t)2 * H3 * HID);
  conv(dec_Wr0,      decWr0_h,   (size_t)2 * HID * HID);
  conv(dec_Wr1,      decWr1_h,   (size_t)2 * HID * HID);
  conv(dec_mlp_W0,   decmlpW0_h, (size_t)HID * H3);
  conv(dec_mlp_W1,   decmlpW1_h, (size_t)FEAT * HID);

  // encoder fc1 -> Hcat cols [0,HID)
  gemm(feat_h, EMB, encfc1W_h, EMB, enc_fc1_b,
       HcatEncF, H3, HcatEncH, H3, ROWS, HID, EMB, 1);

  // encoder dag scans
  float* adjBs[2] = {adjB0, adjB1};
  for (int l = 0; l < 2; ++l) {
    const half_t* Hl_h = HcatEncH + l * HID;
    const float*  Hl_f = HcatEncF + l * HID;
    gemm(Hl_h, H3, encgcWih_h + (size_t)l * H3 * HID, HID, enc_gruc_bih + l * H3,
         Gqc, H3, nullptr, 0, ROWS, H3, HID, 0);
    gemm(Hl_h, H3, encgpWhh_h + (size_t)l * H3 * HID, HID, enc_grup_bhh + l * H3,
         Gqp, H3, nullptr, 0, ROWS, H3, HID, 0);
    scan_kernel<<<1, 768, 0, stream>>>(
        Hl_f, Hl_h, H3,
        HcatEncF + (l + 1) * HID, HcatEncH + (l + 1) * HID, H3,
        Gqc, Gqp, enc_gruc_bhh + l * H3, enc_grup_bih + l * H3,
        encgcWhh_h + (size_t)l * H3 * HID, encgpWih_h + (size_t)l * H3 * HID,
        encWr0_h + (size_t)l * HID * HID, encWr1_h + (size_t)l * HID * HID,
        P0, P1, kbuf,
        enc_gat_w + l * 2 * HID, enc_gat_b + l,
        adj, s_mask, nullptr, adjBs[l], Ghc, Gip, 0);
  }

  // encoder MLP -> fU
  gemm(HcatEncH, H3, encmlpW0_h, H3, enc_mlp_b0, tmpF, HID, tmpH, HID,
       ROWS, HID, H3, 1);
  gemm(tmpH, HID, encmlpW1_h, HID, enc_mlp_b1, fUF, CODE, fUH, CODE,
       ROWS, CODE, HID, 0);

  // decoder fc1
  gemm(fUH, CODE, decfc1W_h, CODE, dec_fc1_b,
       HcatDecF, H3, HcatDecH, H3, ROWS, HID, CODE, 1);

  // decoder dag scans (b_inv = I - adjB)
  for (int l = 0; l < 2; ++l) {
    const half_t* Dl_h = HcatDecH + l * HID;
    const float*  Dl_f = HcatDecF + l * HID;
    gemm(Dl_h, H3, decgcWih_h + (size_t)l * H3 * HID, HID, dec_gruc_bih + l * H3,
         Gqc, H3, nullptr, 0, ROWS, H3, HID, 0);
    gemm(Dl_h, H3, decgpWhh_h + (size_t)l * H3 * HID, HID, dec_grup_bhh + l * H3,
         Gqp, H3, nullptr, 0, ROWS, H3, HID, 0);
    scan_kernel<<<1, 768, 0, stream>>>(
        Dl_f, Dl_h, H3,
        HcatDecF + (l + 1) * HID, HcatDecH + (l + 1) * HID, H3,
        Gqc, Gqp, dec_gruc_bhh + l * H3, dec_grup_bih + l * H3,
        decgcWhh_h + (size_t)l * H3 * HID, decgpWih_h + (size_t)l * H3 * HID,
        decWr0_h + (size_t)l * HID * HID, decWr1_h + (size_t)l * HID * HID,
        P0, P1, kbuf,
        nullptr, nullptr, adj, s_mask, adjBs[l], nullptr, Ghc, Gip, 1);
  }

  // decoder MLP -> logits
  float* logits = (float*)d_out;
  gemm(HcatDecH, H3, decmlpW0_h, H3, dec_mlp_b0, tmpF, HID, tmpH, HID,
       ROWS, HID, H3, 1);
  gemm(tmpH, HID, decmlpW1_h, HID, dec_mlp_b1, logits, FEAT, nullptr, 0,
       ROWS, FEAT, HID, 0);

  // feature_map = I - adjB[1]
  float* fmap = logits + (size_t)ROWS * FEAT;
  feature_map_kernel<<<(BB * NN * NN + 255) / 256, 256, 0, stream>>>(adjB1, fmap);
}
